// LayoutTransformerEncoder_69114613727623
// MI455X (gfx1250) — matrix operand order repaired and checked by reading, not compiled, not run on hardware
//
#include <hip/hip_runtime.h>
#include <hip/hip_bf16.h>
#include <math.h>
#include <cstdint>

// ---------------------------------------------------------------------------
// Types for CDNA5 WMMA (wave32): D = A(16x32 bf16) * B(32x16 bf16) + C(f32)
// ---------------------------------------------------------------------------
typedef __bf16 bf16_t;
typedef __attribute__((ext_vector_type(4)))  bf16_t v4bf;
typedef __attribute__((ext_vector_type(8)))  bf16_t v8bf;
typedef __attribute__((ext_vector_type(16))) bf16_t v16bf;
typedef __attribute__((ext_vector_type(8)))  float  v8f;

static __device__ inline v8f zero8() {
  v8f z;
#pragma unroll
  for (int i = 0; i < 8; ++i) z[i] = 0.0f;
  return z;
}

static __device__ inline v16bf cat8(v8bf lo, v8bf hi) {
  return __builtin_shufflevector(lo, hi, 0, 1, 2, 3, 4, 5, 6, 7,
                                 8, 9, 10, 11, 12, 13, 14, 15);
}

// A-fragment (16x32 bf16): elements 0..15 = K[b..b+15] (contiguous).
static __device__ inline v16bf afrag(const bf16_t* p) {
  v8bf lo = *(const v8bf*)(p);
  v8bf hi = *(const v8bf*)(p + 8);
  return cat8(lo, hi);
}

// B-fragment (32x16 bf16): lane holds col n = lane&15, 16 contiguous K.
static __device__ inline v16bf bfrag(const bf16_t* p) {
  v8bf lo = *(const v8bf*)(p);
  v8bf hi = *(const v8bf*)(p + 8);
  return cat8(lo, hi);
}

// B fragment from fp32 global (used only by the tiny final projection).
static __device__ inline v16bf bfrag_f32(const float* p) {
  const float4* q = (const float4*)p;
  float4 a0 = q[0], a1 = q[1], a2 = q[2], a3 = q[3];
  float f[16] = {a0.x, a0.y, a0.z, a0.w, a1.x, a1.y, a1.z, a1.w,
                 a2.x, a2.y, a2.z, a2.w, a3.x, a3.y, a3.z, a3.w};
  v16bf r;
#pragma unroll
  for (int i = 0; i < 16; ++i) r[i] = (bf16_t)f[i];
  return r;
}

static __device__ inline v8f wmma_bf16(v16bf a, v16bf b, v8f c) {
  return __builtin_amdgcn_wmma_f32_16x16x32_bf16(
      false, a, false, b, (short)0, c, false, false);
}

// ---------------------------------------------------------------------------
// CDNA5 async global->LDS copy (ASYNCcnt-tracked), 16 bytes per lane.
// vdst = LDS byte address (low 32 bits of the generic shared pointer),
// vaddr = 64-bit global address, no SADDR.
// ---------------------------------------------------------------------------
static __device__ inline void async_ld16(const bf16_t* gsrc, uint32_t lds_off) {
  asm volatile("global_load_async_to_lds_b128 %0, %1, off"
               :
               : "v"(lds_off), "v"(gsrc)
               : "memory");
}
static __device__ inline void wait_async_le8() {
  asm volatile("s_wait_asynccnt 0x8" ::: "memory");
}
static __device__ inline void wait_async_le0() {
  asm volatile("s_wait_asynccnt 0x0" ::: "memory");
}

static __device__ inline void store_elt(float* p, float v) { *p = v; }
static __device__ inline void store_elt(bf16_t* p, float v) { *p = (bf16_t)v; }

// ---------------------------------------------------------------------------
// bf16 GEMM: C[M,N] = act(A[M,K] @ W[N,K]^T + bias) (+Res)
// A, W are bf16 in global memory. Block = 256 threads = 8 waves.
// Block tile 128(M) x 128(N); wave = 16 rows x 128 cols (8 accumulators).
// K staged in chunks of 64 via async LDS copies, double-buffered on ASYNCcnt.
// ---------------------------------------------------------------------------
template <int ACT, bool RES, typename OT>
__global__ __launch_bounds__(256) void gemm_bf16_kernel(
    const bf16_t* __restrict__ A, long lda, const bf16_t* __restrict__ W,
    const float* __restrict__ bias, OT* __restrict__ C,
    const float* __restrict__ Res, int M, int N, int K) {
  __shared__ bf16_t sA[2][128 * 72];  // 128 rows x 64 K, padded to 72
  __shared__ bf16_t sW[2][128 * 72];

  const int tid = threadIdx.x;
  const int lane = tid & 31;
  const int wave = tid >> 5;
  const int mBlk = blockIdx.x * 128;
  const int nBlk = blockIdx.y * 128;

  const int acol = lane & 15;
  const int khalf8 = (lane >> 4) << 3;
  const int khalf16 = (lane >> 4) << 4;

  const int nchunks = K >> 6;

  // Stage chunk c into buffer b: 1024 16B segments per operand, 4 per thread.
  auto issue = [&](int c, int b) {
#pragma unroll
    for (int i = 0; i < 4; ++i) {
      const int s = tid + i * 256;
      const int row = s >> 3;
      const int seg = s & 7;
      const bf16_t* ga = A + (long)(mBlk + row) * lda + c * 64 + seg * 8;
      async_ld16(ga, (uint32_t)(uintptr_t)&sA[b][row * 72 + seg * 8]);
      const bf16_t* gw = W + (long)(nBlk + row) * K + c * 64 + seg * 8;
      async_ld16(gw, (uint32_t)(uintptr_t)&sW[b][row * 72 + seg * 8]);
    }
  };

  v8f acc[8];
#pragma unroll
  for (int t = 0; t < 8; ++t) acc[t] = zero8();

  issue(0, 0);
  for (int c = 0; c < nchunks; ++c) {
    if (c + 1 < nchunks) {
      issue(c + 1, (c + 1) & 1);
      wait_async_le8();  // chunk c complete; chunk c+1 still in flight
    } else {
      wait_async_le0();
    }
    __syncthreads();

    const bf16_t* bufA = sA[c & 1];
    const bf16_t* bufW = sW[c & 1];
#pragma unroll
    for (int ks = 0; ks < 2; ++ks) {
      v16bf a = afrag(&bufA[(wave * 16 + acol) * 72 + ks * 32 + khalf16]);
#pragma unroll
      for (int t = 0; t < 8; ++t) {
        v16bf b = bfrag(&bufW[(t * 16 + acol) * 72 + ks * 32 + khalf16]);
        acc[t] = wmma_bf16(a, b, acc[t]);
      }
    }
    __syncthreads();
  }

#pragma unroll
  for (int t = 0; t < 8; ++t) {
    const int col = nBlk + t * 16 + acol;
    const float bv = bias ? bias[col] : 0.0f;
#pragma unroll
    for (int r = 0; r < 8; ++r) {
      const int row = mBlk + wave * 16 + r + khalf8;
      float v = acc[t][r] + bv;
      if (ACT == 1) v = 0.5f * v * (1.0f + erff(v * 0.70710678118654752f));
      const long idx = (long)row * N + col;
      if (RES) v += Res[idx];
      store_elt(&C[idx], v);
    }
  }
}

// ---------------------------------------------------------------------------
// fp32-input GEMM (tiny final projection only): 16x128 block tile.
// ---------------------------------------------------------------------------
__global__ __launch_bounds__(256) void gemm_f32_kernel(
    const float* __restrict__ A, long lda, const float* __restrict__ W,
    const float* __restrict__ bias, float* __restrict__ C, int M, int N,
    int K) {
  __shared__ bf16_t aLds[16 * 136];

  const int tid = threadIdx.x;
  const int lane = tid & 31;
  const int wave = tid >> 5;
  const int m0 = blockIdx.x * 16;
  const int n0 = blockIdx.y * 128 + wave * 16;

  const int srow = tid & 15;
  const int sseg = tid >> 4;
  const int acol = lane & 15;
  const int khalf8 = (lane >> 4) << 3;
  const int khalf16 = (lane >> 4) << 4;

  v8f acc = zero8();
  const float* wrow = W + (long)(n0 + acol) * K + khalf16;

  for (int kc = 0; kc < K; kc += 128) {
    const float* ap = A + (long)(m0 + srow) * lda + kc + sseg * 8;
    float4 f0 = ((const float4*)ap)[0];
    float4 f1 = ((const float4*)ap)[1];
    v8bf pk;
    pk[0] = (bf16_t)f0.x; pk[1] = (bf16_t)f0.y;
    pk[2] = (bf16_t)f0.z; pk[3] = (bf16_t)f0.w;
    pk[4] = (bf16_t)f1.x; pk[5] = (bf16_t)f1.y;
    pk[6] = (bf16_t)f1.z; pk[7] = (bf16_t)f1.w;
    *(v8bf*)&aLds[srow * 136 + sseg * 8] = pk;
    __syncthreads();
#pragma unroll
    for (int ks = 0; ks < 4; ++ks) {
      v16bf a = afrag(&aLds[acol * 136 + ks * 32 + khalf16]);
      v16bf b = bfrag_f32(wrow + kc + ks * 32);
      acc = wmma_bf16(a, b, acc);
    }
    __syncthreads();
  }

  const int col = n0 + acol;
  const float bv = bias ? bias[col] : 0.0f;
#pragma unroll
  for (int r = 0; r < 8; ++r) {
    const int row = m0 + r + khalf8;
    C[(long)row * N + col] = acc[r] + bv;
  }
}

// ---------------------------------------------------------------------------
// Fused multi-head attention, one (batch, head) per block; qkv is bf16.
// scores = (q/8) @ k^T -> softmax -> out = P @ v, all WMMA.
// ---------------------------------------------------------------------------
__global__ __launch_bounds__(256) void attn_wmma_kernel(
    const bf16_t* __restrict__ qkv, bf16_t* __restrict__ aout) {
  __shared__ bf16_t smem[18432 + 8704];
  bf16_t* q_s = smem;          // [128][72]
  bf16_t* k_s = smem + 9216;   // [128][72]
  bf16_t* vT = smem + 18432;   // [64][136] transposed [c][s]
  bf16_t* p_s = smem;          // [128][136] probs, overlays q/k after barrier

  const int bh = blockIdx.x;
  const int b = bh >> 3;
  const int h = bh & 7;
  const bf16_t* base = qkv + (long)b * 128 * 1536 + h * 192;

  const int tid = threadIdx.x;
  for (int i = tid; i < 128 * 64; i += 256) {
    const int s = i >> 6;
    const int c = i & 63;
    const bf16_t* p = base + (long)s * 1536 + c;
    q_s[s * 72 + c] = (bf16_t)((float)p[0] * 0.125f);  // scale^2 folded in
    k_s[s * 72 + c] = p[64];
    vT[c * 136 + s] = p[128];
  }
  __syncthreads();

  const int lane = tid & 31;
  const int wave = tid >> 5;
  const int m0 = wave * 16;
  const int acol = lane & 15;
  const int khalf8 = (lane >> 4) << 3;
  const int khalf16 = (lane >> 4) << 4;

  v8f sc[8];
#pragma unroll
  for (int t = 0; t < 8; ++t) sc[t] = zero8();
#pragma unroll
  for (int ks = 0; ks < 2; ++ks) {
    v16bf a = afrag(&q_s[(m0 + acol) * 72 + ks * 32 + khalf16]);
#pragma unroll
    for (int t = 0; t < 8; ++t) {
      v16bf bf = bfrag(&k_s[(t * 16 + acol) * 72 + ks * 32 + khalf16]);
      sc[t] = wmma_bf16(a, bf, sc[t]);
    }
  }
  __syncthreads();  // q_s/k_s dead; p_s may overlay

#pragma unroll
  for (int r = 0; r < 8; ++r) {
    float mx = -3.4e38f;
#pragma unroll
    for (int t = 0; t < 8; ++t) mx = fmaxf(mx, sc[t][r]);
#pragma unroll
    for (int off = 8; off >= 1; off >>= 1)
      mx = fmaxf(mx, __shfl_xor(mx, off, 32));
    float sum = 0.0f;
#pragma unroll
    for (int t = 0; t < 8; ++t) {
      const float e = __expf(sc[t][r] - mx);
      sc[t][r] = e;
      sum += e;
    }
#pragma unroll
    for (int off = 8; off >= 1; off >>= 1) sum += __shfl_xor(sum, off, 32);
    const float inv = 1.0f / sum;
    const int row = m0 + r + khalf8;
#pragma unroll
    for (int t = 0; t < 8; ++t)
      p_s[row * 136 + t * 16 + acol] = (bf16_t)(sc[t][r] * inv);
  }

#pragma unroll
  for (int nt = 0; nt < 4; ++nt) {
    v8f acc = zero8();
#pragma unroll
    for (int ks = 0; ks < 4; ++ks) {
      v16bf a = afrag(&p_s[(m0 + acol) * 136 + ks * 32 + khalf16]);
      v16bf bf = bfrag(&vT[(nt * 16 + acol) * 136 + ks * 32 + khalf16]);
      acc = wmma_bf16(a, bf, acc);
    }
#pragma unroll
    for (int r = 0; r < 8; ++r) {
      const int row = m0 + r + khalf8;
      aout[((long)(b * 128 + row)) * 512 + h * 64 + nt * 16 + acol] =
          (bf16_t)acc[r];
    }
  }
}

// ---------------------------------------------------------------------------
// LayerNorm: one wave per token, D = 512. OT selects f32 or bf16 output.
// ---------------------------------------------------------------------------
static __device__ inline void store4(float* p, float4 o) { *(float4*)p = o; }
static __device__ inline void store4(bf16_t* p, float4 o) {
  v4bf v;
  v[0] = (bf16_t)o.x; v[1] = (bf16_t)o.y;
  v[2] = (bf16_t)o.z; v[3] = (bf16_t)o.w;
  *(v4bf*)p = v;
}

template <typename OT>
__global__ __launch_bounds__(256) void ln_kernel(const float* __restrict__ x,
                                                 const float* __restrict__ g,
                                                 const float* __restrict__ bta,
                                                 OT* __restrict__ out) {
  const int lane = threadIdx.x & 31;
  const int wave = threadIdx.x >> 5;
  const long tok = (long)blockIdx.x * 8 + wave;
  const float* xp = x + tok * 512;

  float4 v[4];
  float s = 0.0f, sq = 0.0f;
#pragma unroll
  for (int i = 0; i < 4; ++i) {
    v[i] = ((const float4*)xp)[i * 32 + lane];
    s += v[i].x + v[i].y + v[i].z + v[i].w;
    sq += v[i].x * v[i].x + v[i].y * v[i].y + v[i].z * v[i].z + v[i].w * v[i].w;
  }
#pragma unroll
  for (int off = 16; off >= 1; off >>= 1) {
    s += __shfl_xor(s, off, 32);
    sq += __shfl_xor(sq, off, 32);
  }
  const float mean = s * (1.0f / 512.0f);
  const float var = sq * (1.0f / 512.0f) - mean * mean;
  const float rs = rsqrtf(var + 1e-5f);

  OT* op = out + tok * 512;
#pragma unroll
  for (int i = 0; i < 4; ++i) {
    float4 gg = ((const float4*)g)[i * 32 + lane];
    float4 bb = ((const float4*)bta)[i * 32 + lane];
    float4 o;
    o.x = (v[i].x - mean) * rs * gg.x + bb.x;
    o.y = (v[i].y - mean) * rs * gg.y + bb.y;
    o.z = (v[i].z - mean) * rs * gg.z + bb.z;
    o.w = (v[i].w - mean) * rs * gg.w + bb.w;
    store4(op + i * 128 + lane * 4, o);
  }
}

// ---------------------------------------------------------------------------
// fp32 -> bf16 conversion (weights / mask input); n must be multiple of 1024.
// ---------------------------------------------------------------------------
__global__ __launch_bounds__(256) void cvt_bf16_kernel(
    const float* __restrict__ in, bf16_t* __restrict__ out) {
  const long i = (long)blockIdx.x * 1024 + threadIdx.x * 4;
  float4 f = *(const float4*)(in + i);
  v4bf o;
  o[0] = (bf16_t)f.x; o[1] = (bf16_t)f.y;
  o[2] = (bf16_t)f.z; o[3] = (bf16_t)f.w;
  *(v4bf*)(out + i) = o;
}

// ---------------------------------------------------------------------------
// Embedding base: x = pos + class_table[cls] + bbox @ bbox_w^T + bbox_b + mb
// ---------------------------------------------------------------------------
__global__ __launch_bounds__(256) void embed_base_kernel(
    const int* __restrict__ cls, const float* __restrict__ bbox,
    const float* __restrict__ pos, const float* __restrict__ ctab,
    const float* __restrict__ bw, const float* __restrict__ bb,
    const float* __restrict__ mb, float* __restrict__ x) {
  const long tok = blockIdx.x;
  const int l = (int)(tok & 127);
  const int c = cls[tok];
  const float b0 = bbox[tok * 4 + 0];
  const float b1 = bbox[tok * 4 + 1];
  const float b2 = bbox[tok * 4 + 2];
  const float b3 = bbox[tok * 4 + 3];
  for (int d = threadIdx.x; d < 512; d += 256) {
    float v = pos[l * 512 + d] + ctab[(long)c * 512 + d] + bb[d] + mb[d] +
              b0 * bw[d * 4 + 0] + b1 * bw[d * 4 + 1] + b2 * bw[d * 4 + 2] +
              b3 * bw[d * 4 + 3];
    x[tok * 512 + d] = v;
  }
}

// ---------------------------------------------------------------------------
// Pack output: out[b, d, 0] = proj[b, d]; out[b, d, 1+l] = xf[b, l, d]
// ---------------------------------------------------------------------------
__global__ __launch_bounds__(256) void pack_kernel(
    const float* __restrict__ xf, const float* __restrict__ proj,
    float* __restrict__ out) {
  const int b = blockIdx.x;
  for (int i = threadIdx.x; i < 512 * 129; i += 256) {
    const int d = i / 129;
    const int col = i % 129;
    float v = (col == 0) ? proj[b * 512 + d]
                         : xf[((long)(b * 128 + col - 1)) * 512 + d];
    out[(long)b * 512 * 129 + i] = v;
  }
}

// ---------------------------------------------------------------------------
// Host-side orchestration
// ---------------------------------------------------------------------------
extern "C" void kernel_launch(void* const* d_in, const int* in_sizes, int n_in,
                              void* d_out, int out_size, void* d_ws,
                              size_t ws_size, hipStream_t stream) {
  const int* obj_class = (const int*)d_in[0];
  const float* obj_bbox = (const float*)d_in[1];
  const float* obj_mask = (const float*)d_in[2];
  const float* pos_emb = (const float*)d_in[3];
  const float* class_table = (const float*)d_in[4];
  const float* bbox_w = (const float*)d_in[5];
  const float* bbox_b = (const float*)d_in[6];
  const float* mask_w = (const float*)d_in[7];
  const float* mask_b = (const float*)d_in[8];
  const float* ln1_g = (const float*)d_in[9];
  const float* ln1_b = (const float*)d_in[10];
  const float* qkv_w = (const float*)d_in[11];
  const float* qkv_b = (const float*)d_in[12];
  const float* attn_w = (const float*)d_in[13];
  const float* attn_b = (const float*)d_in[14];
  const float* ln2_g = (const float*)d_in[15];
  const float* ln2_b = (const float*)d_in[16];
  const float* fc_w = (const float*)d_in[17];
  const float* fc_b = (const float*)d_in[18];
  const float* mproj_w = (const float*)d_in[19];
  const float* mproj_b = (const float*)d_in[20];
  const float* fln_g = (const float*)d_in[21];
  const float* fln_b = (const float*)d_in[22];
  const float* out_w = (const float*)d_in[23];
  const float* out_b = (const float*)d_in[24];
  float* out = (float*)d_out;

  const int BL = 256 * 128;  // 32768 tokens
  const size_t MB = 1024 * 1024;

  // Workspace layout:
  char* ws = (char*)d_ws;
  float* x = (float*)(ws);                       // [BL,512] f32, 64MB
  bf16_t* hb = (bf16_t*)(ws + 64 * MB);          // [BL,512] bf16, 32MB
  bf16_t* ab = hb;                               // attn out aliases LN buffer
  bf16_t* qkvb = (bf16_t*)(ws + 96 * MB);        // [BL,1536] bf16 (96MB)
  bf16_t* fb = qkvb;                             // [BL,2048] bf16 (128MB max)
  float* hf = (float*)(ws + 96 * MB);            // final LN f32, aliases qkvb
  size_t off = 224 * MB;
  bf16_t* qkv_wb = (bf16_t*)(ws + off);  off += (size_t)6 * 1536 * 512 * 2;
  bf16_t* attn_wb = (bf16_t*)(ws + off); off += (size_t)6 * 512 * 512 * 2;
  bf16_t* fc_wb = (bf16_t*)(ws + off);   off += (size_t)6 * 2048 * 512 * 2;
  bf16_t* mproj_wb = (bf16_t*)(ws + off); off += (size_t)6 * 512 * 2048 * 2;
  bf16_t* mask_wb = (bf16_t*)(ws + off); off += (size_t)512 * 1024 * 2;
  bf16_t* maskb = (bf16_t*)(ws + off);   off += (size_t)BL * 1024 * 2;
  float* projb = (float*)(ws + off);     off += (size_t)256 * 512 * 4;

  // One-time (per call) bf16 conversions of weights + mask features.
  cvt_bf16_kernel<<<6 * 1536 * 512 / 1024, 256, 0, stream>>>(qkv_w, qkv_wb);
  cvt_bf16_kernel<<<6 * 512 * 512 / 1024, 256, 0, stream>>>(attn_w, attn_wb);
  cvt_bf16_kernel<<<6 * 2048 * 512 / 1024, 256, 0, stream>>>(fc_w, fc_wb);
  cvt_bf16_kernel<<<6 * 512 * 2048 / 1024, 256, 0, stream>>>(mproj_w, mproj_wb);
  cvt_bf16_kernel<<<512 * 1024 / 1024, 256, 0, stream>>>(mask_w, mask_wb);
  cvt_bf16_kernel<<<BL * 1024 / 1024, 256, 0, stream>>>(obj_mask, maskb);

  // Embedding: base terms, then x += mask @ mask_w^T via WMMA GEMM.
  embed_base_kernel<<<BL, 256, 0, stream>>>(obj_class, obj_bbox, pos_emb,
                                            class_table, bbox_w, bbox_b,
                                            mask_b, x);
  gemm_bf16_kernel<0, true, float>
      <<<dim3(BL / 128, 512 / 128), 256, 0, stream>>>(
          maskb, 1024, mask_wb, nullptr, x, x, BL, 512, 1024);

  for (int i = 0; i < 6; ++i) {
    ln_kernel<bf16_t><<<BL / 8, 256, 0, stream>>>(x, ln1_g + i * 512,
                                                  ln1_b + i * 512, hb);
    gemm_bf16_kernel<0, false, bf16_t>
        <<<dim3(BL / 128, 1536 / 128), 256, 0, stream>>>(
            hb, 512, qkv_wb + (size_t)i * 1536 * 512, qkv_b + i * 1536, qkvb,
            nullptr, BL, 1536, 512);
    attn_wmma_kernel<<<256 * 8, 256, 0, stream>>>(qkvb, ab);
    gemm_bf16_kernel<0, true, float>
        <<<dim3(BL / 128, 512 / 128), 256, 0, stream>>>(
            ab, 512, attn_wb + (size_t)i * 512 * 512, attn_b + i * 512, x, x,
            BL, 512, 512);
    ln_kernel<bf16_t><<<BL / 8, 256, 0, stream>>>(x, ln2_g + i * 512,
                                                  ln2_b + i * 512, hb);
    gemm_bf16_kernel<1, false, bf16_t>
        <<<dim3(BL / 128, 2048 / 128), 256, 0, stream>>>(
            hb, 512, fc_wb + (size_t)i * 2048 * 512, fc_b + i * 2048, fb,
            nullptr, BL, 2048, 512);
    gemm_bf16_kernel<0, true, float>
        <<<dim3(BL / 128, 512 / 128), 256, 0, stream>>>(
            fb, 2048, mproj_wb + (size_t)i * 512 * 2048, mproj_b + i * 512, x,
            x, BL, 512, 2048);
  }

  // Final LN (f32 output: it IS part of the result), tiny projection, pack.
  ln_kernel<float><<<BL / 8, 256, 0, stream>>>(x, fln_g, fln_b, hf);
  gemm_f32_kernel<<<dim3(256 / 16, 512 / 128), 256, 0, stream>>>(
      hf, (long)128 * 512, out_w, out_b, projb, 256, 512, 512);
  pack_kernel<<<256, 256, 0, stream>>>(hf, projb, out);
}
